// PolicyModel_60404420051142
// MI455X (gfx1250) — matrix operand order repaired and checked by reading, not compiled, not run on hardware
//
#include <hip/hip_runtime.h>
#include <cstdint>

// ---------------------------------------------------------------------------
// Problem constants (match reference)
// ---------------------------------------------------------------------------
static constexpr int NVERT = 5000;
static constexpr int FDIM  = 7699;
static constexpr int EDGES = 40000;
static constexpr int H1    = 4;
static constexpr int C1    = 1024;
static constexpr int HC1   = H1 * C1;     // 4096
static constexpr int C2    = 1024;
static constexpr int NOUT  = 128;
// Padded GEMM dims (MP mult of 128, KP1 mult of 64) so the hot loop is guard-free
static constexpr int MP    = 5120;        // padded node count
static constexpr int KP1   = 7744;        // padded FDIM (mult of 64)

typedef unsigned short u16;
typedef __attribute__((ext_vector_type(4)))  unsigned int u32x4;
typedef __attribute__((ext_vector_type(4)))  int          i32x4;
typedef __attribute__((ext_vector_type(16))) __bf16       bf16x16;
typedef __attribute__((ext_vector_type(8)))  float        f32x8;

#define AS1 __attribute__((address_space(1)))
#define AS3 __attribute__((address_space(3)))

union FragAB { bf16x16 v; u32x4 q[2]; };

// CDNA5 async global->LDS DMA (ASYNCcnt-tracked), with fallback
#if defined(__has_builtin)
#  if __has_builtin(__builtin_amdgcn_global_load_async_to_lds_b128) && \
      __has_builtin(__builtin_amdgcn_s_wait_asynccnt)
#    define ASYNC_LDS 1
#  endif
#endif
#ifndef ASYNC_LDS
#  define ASYNC_LDS 0
#endif

__device__ __forceinline__ u16 f32_to_bf16(float f) {
    unsigned int u = __float_as_uint(f);
    u += 0x7FFFu + ((u >> 16) & 1u);      // round to nearest even
    return (u16)(u >> 16);
}

// ---------------------------------------------------------------------------
// f32 -> bf16 convert with zero padding: dst is [Rp x Cp], src is [R x C]
// ---------------------------------------------------------------------------
__global__ void k_cvt(const float* __restrict__ src, u16* __restrict__ dst,
                      int R, int C, long long Rp, int Cp) {
    long long i = (long long)blockIdx.x * blockDim.x + threadIdx.x;
    long long tot = Rp * (long long)Cp;
    if (i >= tot) return;
    int c = (int)(i % Cp);
    long long r = i / Cp;
    float v = (r < R && c < C) ? src[r * (long long)C + c] : 0.0f;
    dst[i] = f32_to_bf16(v);
}

// ---------------------------------------------------------------------------
// WMMA bf16 GEMM:  C[M,N] = A[MP,Kp] * B[Kp,N] + bias[N]
// 256 threads = 8 wave32s; block tile 128x128; K-step 64; double-buffered LDS;
// A tile staged via async global->LDS DMA (ASYNCcnt), B staged K-transposed
// through registers. Next tile's loads are issued before this tile's WMMAs.
// ---------------------------------------------------------------------------
#define BM 128
#define BN 128
#define BK 64
#define LDK 72   // padded LDS row: 144B stride (multiple of 16B)

__global__ __launch_bounds__(256) void k_gemm_bf16(
    const u16* __restrict__ A, const u16* __restrict__ B,
    const float* __restrict__ bias, float* __restrict__ C,
    int M, int Nn, int Kp)
{
    __shared__ __align__(16) u16 As[2][BM][LDK];   // [buf][m][k]
    __shared__ __align__(16) u16 Bs[2][BN][LDK];   // [buf][n][k] (K-transposed)

    const int tid  = threadIdx.x;
    const int lane = tid & 31;
    const int wave = tid >> 5;
    const int wm   = wave >> 2;      // 0..1
    const int wn   = wave & 3;       // 0..3
    const int half = lane >> 4;      // 0/1
    const int l16  = lane & 15;
    const int m0   = blockIdx.y * BM;
    const int n0   = blockIdx.x * BN;

    f32x8 acc[4][2];
    const f32x8 zero = {0.f,0.f,0.f,0.f,0.f,0.f,0.f,0.f};
#pragma unroll
    for (int mi = 0; mi < 4; ++mi) { acc[mi][0] = zero; acc[mi][1] = zero; }

    u32x4 Breg[4];
#if !ASYNC_LDS
    u32x4 Areg[4];
#endif

    // ---- A tile (128x64): 1024 x 16B chunks, 4 per thread ----
    auto stageA_issue = [&](int k0, int buf) {
#pragma unroll
        for (int j = 0; j < 4; ++j) {
            int i = tid + j * 256;
            int r = i >> 3, cs = i & 7;
            const u16* gp = A + (size_t)(m0 + r) * Kp + k0 + cs * 8;
#if ASYNC_LDS
            u16* lp = &As[buf][r][cs * 8];
            __builtin_amdgcn_global_load_async_to_lds_b128(
                (AS1 i32x4*)(AS1 void*)const_cast<u16*>(gp),
                (AS3 i32x4*)(AS3 void*)lp, 0, 0);
#else
            Areg[j] = *(const u32x4*)gp;
            (void)buf;
#endif
        }
    };
    auto stageA_store = [&](int buf) {
#if !ASYNC_LDS
#pragma unroll
        for (int j = 0; j < 4; ++j) {
            int i = tid + j * 256;
            int r = i >> 3, cs = i & 7;
            *(u32x4*)&As[buf][r][cs * 8] = Areg[j];
        }
#else
        (void)buf;
#endif
    };
    // ---- B tile (64x128): coalesced 16B global loads, transposed LDS stores ----
    auto loadB = [&](int k0) {
#pragma unroll
        for (int j = 0; j < 4; ++j) {
            int i = tid + j * 256;
            int kk = i >> 4, ng = i & 15;
            Breg[j] = *(const u32x4*)(B + (size_t)(k0 + kk) * Nn + n0 + ng * 8);
        }
    };
    auto storeB = [&](int buf) {
#pragma unroll
        for (int j = 0; j < 4; ++j) {
            int i = tid + j * 256;
            int kk = i >> 4, ng = i & 15;
            const u16* pv = (const u16*)&Breg[j];
#pragma unroll
            for (int t = 0; t < 8; ++t) Bs[buf][ng * 8 + t][kk] = pv[t];
        }
    };
    // ---- 16 WMMAs on one buffered K-tile ----
    auto compute = [&](int buf) {
#pragma unroll
        for (int ks = 0; ks < BK; ks += 32) {
            FragAB a[4], b[2];
#pragma unroll
            for (int mi = 0; mi < 4; ++mi) {
                const u16* pr = &As[buf][wm * 64 + mi * 16 + l16][ks];
                a[mi].q[0] = *(const u32x4*)(pr + half * 8);        // K 0..7 / 8..15
                a[mi].q[1] = *(const u32x4*)(pr + 16 + half * 8);   // K 16..23 / 24..31
            }
#pragma unroll
            for (int ni = 0; ni < 2; ++ni) {
                const u16* pr = &Bs[buf][wn * 32 + ni * 16 + l16][ks];
                b[ni].q[0] = *(const u32x4*)(pr + half * 8);        // K 0..7 / 8..15
                b[ni].q[1] = *(const u32x4*)(pr + 16 + half * 8);   // K 16..23 / 24..31
            }
#pragma unroll
            for (int mi = 0; mi < 4; ++mi)
#pragma unroll
                for (int ni = 0; ni < 2; ++ni)
                    acc[mi][ni] = __builtin_amdgcn_wmma_f32_16x16x32_bf16(
                        false, a[mi].v, false, b[ni].v,
                        (short)0, acc[mi][ni], false, false);
        }
    };

    const int nt = Kp / BK;

    // prologue: stage tile 0 into buffer 0
    stageA_issue(0, 0);
    loadB(0);
    storeB(0);
    stageA_store(0);
#if ASYNC_LDS
    __builtin_amdgcn_s_wait_asynccnt(0);
#endif
    __syncthreads();

    for (int it = 0; it < nt; ++it) {
        const int c = it & 1;
        const bool more = (it + 1) < nt;
        if (more) {                       // issue next tile's loads early
            stageA_issue((it + 1) * BK, 1 - c);
            loadB((it + 1) * BK);
        }
        compute(c);                       // WMMAs overlap in-flight loads
        if (more) {
            storeB(1 - c);
            stageA_store(1 - c);
#if ASYNC_LDS
            __builtin_amdgcn_s_wait_asynccnt(0);
#endif
        }
        __syncthreads();
    }

    // ---- store C + bias (VGPR r: M = r + 8*half, N = l16) ----
#pragma unroll
    for (int mi = 0; mi < 4; ++mi)
#pragma unroll
        for (int ni = 0; ni < 2; ++ni) {
            int nn2 = n0 + wn * 32 + ni * 16 + l16;
            float bv = bias[nn2];
#pragma unroll
            for (int r = 0; r < 8; ++r) {
                int mm = m0 + wm * 64 + mi * 16 + half * 8 + r;
                if (mm < M) C[(size_t)mm * Nn + nn2] = acc[mi][ni][r] + bv;
            }
        }
}

// ---------------------------------------------------------------------------
// Graph helpers
// ---------------------------------------------------------------------------
__global__ void k_zero_f(float* p, int n) {
    int i = blockIdx.x * blockDim.x + threadIdx.x;
    if (i < n) p[i] = 0.f;
}
__global__ void k_zero_i(int* p, int n) {
    int i = blockIdx.x * blockDim.x + threadIdx.x;
    if (i < n) p[i] = 0;
}
__global__ void k_deg(const int* __restrict__ src, const int* __restrict__ dst,
                      const float* __restrict__ ea, float* deg, float* lattr, int E) {
    (void)src;
    int e = blockIdx.x * blockDim.x + threadIdx.x;
    if (e >= E) return;
    int d = dst[e];
    atomicAdd(&deg[d], 1.0f);
    atomicAdd(&lattr[d * 3 + 0], ea[e * 3 + 0]);
    atomicAdd(&lattr[d * 3 + 1], ea[e * 3 + 1]);
    atomicAdd(&lattr[d * 3 + 2], ea[e * 3 + 2]);
}
__global__ void k_loopdiv(float* lattr, const float* deg, int N) {
    int i = blockIdx.x * blockDim.x + threadIdx.x;
    if (i >= N * 3) return;
    lattr[i] = lattr[i] / fmaxf(deg[i / 3], 1.0f);
}
__global__ void k_count(const int* __restrict__ dst, int* cnt, int E, int N) {
    int e = blockIdx.x * blockDim.x + threadIdx.x;
    if (e >= E + N) return;
    int d = (e < E) ? dst[e] : e - E;
    atomicAdd(&cnt[d], 1);
}
__global__ void k_scan(const int* __restrict__ cnt, int* rowptr, int n) {
    __shared__ int s[1024];
    __shared__ int carry;
    int tid = threadIdx.x;
    if (tid == 0) carry = 0;
    __syncthreads();
    for (int base = 0; base < n; base += 1024) {
        int v = (base + tid < n) ? cnt[base + tid] : 0;
        s[tid] = v;
        __syncthreads();
        for (int off = 1; off < 1024; off <<= 1) {
            int t = (tid >= off) ? s[tid - off] : 0;
            __syncthreads();
            s[tid] += t;
            __syncthreads();
        }
        if (base + tid < n) rowptr[base + tid] = carry + s[tid] - v;
        __syncthreads();
        if (tid == 0) carry += s[1023];
        __syncthreads();
    }
    if (tid == 0) rowptr[n] = carry;
}
__global__ void k_scatter(const int* __restrict__ dst, const int* __restrict__ rowptr,
                          int* fill, int* eid, int E, int N) {
    int e = blockIdx.x * blockDim.x + threadIdx.x;
    if (e >= E + N) return;
    int d = (e < E) ? dst[e] : e - E;
    int pos = atomicAdd(&fill[d], 1);
    eid[rowptr[d] + pos] = e;
}

// ---------------------------------------------------------------------------
// GATv2 edge logits: one wave32 per (edge', head)
// logit = sum_c att[h,c] * leaky_relu(xl[src,h,c] + xr[dst,h,c] + (ea@We)[h,c])
// ---------------------------------------------------------------------------
__global__ __launch_bounds__(256) void k_logits(
    const float* __restrict__ XL, const float* __restrict__ XR,
    const float* __restrict__ eattr, const float* __restrict__ lattr,
    const float* __restrict__ We, const float* __restrict__ att,
    const int* __restrict__ src, const int* __restrict__ dst,
    float* __restrict__ logit, int E, int N, int H, int C)
{
    int gw   = (blockIdx.x * blockDim.x + threadIdx.x) >> 5;
    int lane = threadIdx.x & 31;
    int tot  = (E + N) * H;
    if (gw >= tot) return;
    int e = gw / H;
    int h = gw - e * H;
    int s, d; float a0, a1, a2;
    if (e < E) {
        s = src[e]; d = dst[e];
        a0 = eattr[e * 3]; a1 = eattr[e * 3 + 1]; a2 = eattr[e * 3 + 2];
    } else {
        s = d = e - E;
        a0 = lattr[s * 3]; a1 = lattr[s * 3 + 1]; a2 = lattr[s * 3 + 2];
    }
    int HC = H * C;
    const float* xl = XL + (size_t)s * HC + h * C;
    const float* xr = XR + (size_t)d * HC + h * C;
    const float* w0 = We + h * C;          // We[j*HC + h*C + c]
    const float* at = att + h * C;
    float acc = 0.f;
    for (int c = lane; c < C; c += 32) {
        float ee = a0 * w0[c] + a1 * w0[HC + c] + a2 * w0[2 * HC + c];
        float v  = xl[c] + xr[c] + ee;
        v = (v > 0.f) ? v : 0.2f * v;      // leaky_relu, NEG_SLOPE=0.2
        acc += at[c] * v;
    }
    for (int off = 16; off > 0; off >>= 1) acc += __shfl_down(acc, off, 32);
    if (lane == 0) logit[(size_t)e * H + h] = acc;
}

// ---------------------------------------------------------------------------
// Segment softmax + weighted aggregation: one block per (node, head)
// out[n,h,c] = sum_e softmax(logit)_e * XL[src_e,h,c] + bias[h,c]
// ---------------------------------------------------------------------------
__global__ __launch_bounds__(256) void k_aggregate(
    const float* __restrict__ XL, const float* __restrict__ logit,
    const int* __restrict__ rowptr, const int* __restrict__ eid,
    const int* __restrict__ src, const float* __restrict__ bias,
    float* __restrict__ OUT, int E, int H, int C)
{
    int n = blockIdx.x / H;
    int h = blockIdx.x - n * H;
    int tid = threadIdx.x;
    int r0 = rowptr[n], r1 = rowptr[n + 1];
    __shared__ float red[256];

    float m = -3.0e38f;
    for (int j = r0 + tid; j < r1; j += 256)
        m = fmaxf(m, logit[(size_t)eid[j] * H + h]);
    red[tid] = m; __syncthreads();
    for (int s = 128; s > 0; s >>= 1) {
        if (tid < s) red[tid] = fmaxf(red[tid], red[tid + s]);
        __syncthreads();
    }
    float mx = red[0]; __syncthreads();

    float ds = 0.f;
    for (int j = r0 + tid; j < r1; j += 256)
        ds += expf(logit[(size_t)eid[j] * H + h] - mx);
    red[tid] = ds; __syncthreads();
    for (int s = 128; s > 0; s >>= 1) {
        if (tid < s) red[tid] += red[tid + s];
        __syncthreads();
    }
    float den = red[0] + 1e-16f;

    int HC = H * C;
    for (int c = tid; c < C; c += 256) {
        float acc = 0.f;
        for (int j = r0; j < r1; ++j) {
            int e  = eid[j];
            int s2 = (e < E) ? src[e] : e - E;
            float w = expf(logit[(size_t)e * H + h] - mx);
            acc += w * XL[(size_t)s2 * HC + h * C + c];
        }
        OUT[(size_t)n * HC + h * C + c] = acc / den + bias[h * C + c];
    }
}

// ---------------------------------------------------------------------------
// Host orchestration
// ---------------------------------------------------------------------------
static inline int cdiv(long long a, long long b) { return (int)((a + b - 1) / b); }

extern "C" void kernel_launch(void* const* d_in, const int* in_sizes, int n_in,
                              void* d_out, int out_size, void* d_ws, size_t ws_size,
                              hipStream_t stream) {
    (void)in_sizes; (void)n_in; (void)out_size;

    const float* x     = (const float*)d_in[0];
    const int*   ei    = (const int*)  d_in[1];
    const float* eattr = (const float*)d_in[2];
    const float* W1l   = (const float*)d_in[3];  const float* b1l  = (const float*)d_in[4];
    const float* W1r   = (const float*)d_in[5];  const float* b1r  = (const float*)d_in[6];
    const float* W1e   = (const float*)d_in[7];  const float* att1 = (const float*)d_in[8];
    const float* bias1 = (const float*)d_in[9];
    const float* W2l   = (const float*)d_in[10]; const float* b2l  = (const float*)d_in[11];
    const float* W2r   = (const float*)d_in[12]; const float* b2r  = (const float*)d_in[13];
    const float* W2e   = (const float*)d_in[14]; const float* att2 = (const float*)d_in[15];
    const float* bias2 = (const float*)d_in[16];
    const float* Wlin  = (const float*)d_in[17]; const float* blin = (const float*)d_in[18];
    float* out = (float*)d_out;

    const int* src = ei;             // edge_index[0]
    const int* dst = ei + EDGES;     // edge_index[1]

    // ---- workspace bump allocator with aliasing ----
    char* ws = (char*)d_ws;
    size_t off = 0;
    auto take = [&](size_t bytes) -> char* {
        char* p = ws + off;
        off += (bytes + 255) & ~(size_t)255;
        return p;
    };
    u16*   xb     = (u16*)  take((size_t)MP * KP1 * 2);          // 79.3 MB (later: h1b)
    u16*   wb     = (u16*)  take((size_t)KP1 * HC1 * 2);         // 63.4 MB (weights / h2b / wlinb)
    float* XL1    = (float*)take((size_t)NVERT * HC1 * 4);       // 81.9 MB (later: XL2, XR2)
    float* XR1    = (float*)take((size_t)NVERT * HC1 * 4);       // 81.9 MB (later: h1, h2)
    float* deg    = (float*)take((size_t)NVERT * 4);
    float* lattr  = (float*)take((size_t)NVERT * 3 * 4);
    int*   cnt    = (int*)  take((size_t)NVERT * 4);
    int*   rowptr = (int*)  take((size_t)(NVERT + 1) * 4);
    int*   fill   = (int*)  take((size_t)NVERT * 4);
    int*   eid    = (int*)  take((size_t)(EDGES + NVERT) * 4);
    float* logits = (float*)take((size_t)(EDGES + NVERT) * H1 * 4);
    if (off > ws_size) return;   // workspace too small; nothing sane to do

    // aliases (lifetimes are disjoint, kernels run sequentially on `stream`)
    u16*   h1b   = xb;
    float* h1    = XR1;
    u16*   w2lb  = wb;
    u16*   w2rb  = (u16*)((char*)wb + (size_t)(16 << 20));
    float* XL2   = XL1;
    float* XR2   = (float*)((char*)XL1 + (size_t)41943040);
    float* h2    = XR1;
    u16*   h2b   = wb;
    u16*   wlinb = (u16*)((char*)wb + (size_t)12582912);

    const int ET = EDGES + NVERT;

    // ================= Layer 1: GEMMs on WMMA =================
    k_cvt<<<cdiv((long long)MP * KP1, 256), 256, 0, stream>>>(x, xb, NVERT, FDIM, MP, KP1);
    k_cvt<<<cdiv((long long)KP1 * HC1, 256), 256, 0, stream>>>(W1l, wb, FDIM, HC1, KP1, HC1);
    k_gemm_bf16<<<dim3(HC1 / BN, MP / BM), 256, 0, stream>>>(xb, wb, b1l, XL1, NVERT, HC1, KP1);
    k_cvt<<<cdiv((long long)KP1 * HC1, 256), 256, 0, stream>>>(W1r, wb, FDIM, HC1, KP1, HC1);
    k_gemm_bf16<<<dim3(HC1 / BN, MP / BM), 256, 0, stream>>>(xb, wb, b1r, XR1, NVERT, HC1, KP1);

    // ================= Graph prep (shared by both layers) =================
    k_zero_f<<<cdiv(NVERT, 256), 256, 0, stream>>>(deg, NVERT);
    k_zero_f<<<cdiv(NVERT * 3, 256), 256, 0, stream>>>(lattr, NVERT * 3);
    k_zero_i<<<cdiv(NVERT, 256), 256, 0, stream>>>(cnt, NVERT);
    k_zero_i<<<cdiv(NVERT, 256), 256, 0, stream>>>(fill, NVERT);
    k_deg<<<cdiv(EDGES, 256), 256, 0, stream>>>(src, dst, eattr, deg, lattr, EDGES);
    k_loopdiv<<<cdiv(NVERT * 3, 256), 256, 0, stream>>>(lattr, deg, NVERT);
    k_count<<<cdiv(ET, 256), 256, 0, stream>>>(dst, cnt, EDGES, NVERT);
    k_scan<<<1, 1024, 0, stream>>>(cnt, rowptr, NVERT);
    k_scatter<<<cdiv(ET, 256), 256, 0, stream>>>(dst, rowptr, fill, eid, EDGES, NVERT);

    // ================= Layer 1: attention + aggregate =================
    k_logits<<<cdiv((long long)ET * H1 * 32, 256), 256, 0, stream>>>(
        XL1, XR1, eattr, lattr, W1e, att1, src, dst, logits, EDGES, NVERT, H1, C1);
    k_aggregate<<<NVERT * H1, 256, 0, stream>>>(
        XL1, logits, rowptr, eid, src, bias1, h1, EDGES, H1, C1);

    // ================= Layer 2 =================
    k_cvt<<<cdiv((long long)MP * HC1, 256), 256, 0, stream>>>(h1, h1b, NVERT, HC1, MP, HC1);
    k_cvt<<<cdiv((long long)HC1 * C2, 256), 256, 0, stream>>>(W2l, w2lb, HC1, C2, HC1, C2);
    k_gemm_bf16<<<dim3(C2 / BN, MP / BM), 256, 0, stream>>>(h1b, w2lb, b2l, XL2, NVERT, C2, HC1);
    k_cvt<<<cdiv((long long)HC1 * C2, 256), 256, 0, stream>>>(W2r, w2rb, HC1, C2, HC1, C2);
    k_gemm_bf16<<<dim3(C2 / BN, MP / BM), 256, 0, stream>>>(h1b, w2rb, b2r, XR2, NVERT, C2, HC1);

    k_logits<<<cdiv((long long)ET * 1 * 32, 256), 256, 0, stream>>>(
        XL2, XR2, eattr, lattr, W2e, att2, src, dst, logits, EDGES, NVERT, 1, C2);
    k_aggregate<<<NVERT, 256, 0, stream>>>(
        XL2, logits, rowptr, eid, src, bias2, h2, EDGES, 1, C2);

    // ================= Final linear (WMMA, N=128) =================
    k_cvt<<<cdiv((long long)MP * C2, 256), 256, 0, stream>>>(h2, h2b, NVERT, C2, MP, C2);
    k_cvt<<<cdiv((long long)C2 * NOUT, 256), 256, 0, stream>>>(Wlin, wlinb, C2, NOUT, C2, NOUT);
    k_gemm_bf16<<<dim3(NOUT / BN, MP / BM), 256, 0, stream>>>(h2b, wlinb, blin, out, NVERT, NOUT, C2);
}
